// CausalSelfAttention_8186207666878
// MI455X (gfx1250) — compile-verified, matrix operand order checked by reading, not run on hardware
//
#include <hip/hip_runtime.h>
#include <cstdint>

typedef unsigned short u16;
typedef unsigned int   u32;

typedef __attribute__((ext_vector_type(16))) __bf16 v16bf;
typedef __attribute__((ext_vector_type(8)))  float  v8f;

union FragU { uint4 q[2]; v16bf f; };

__device__ __forceinline__ u16 f2bf(float x) {
    u32 u = __float_as_uint(x);
    u32 r = (u + 0x7FFFu + ((u >> 16) & 1u)) >> 16;   // round-to-nearest-even
    return (u16)r;
}
__device__ __forceinline__ float bf2f(u16 s) { return __uint_as_float(((u32)s) << 16); }

// Load one 16x32 bf16 WMMA A/B fragment. `p` must already be the lane-resolved
// address: row_base + kbase + g*8 (elements). Elements 0..7 <- k+0..7,
// elements 8..15 <- k+16..23 per the CDNA5 16-bit A-matrix layout.
__device__ __forceinline__ v16bf load_frag16(const u16* p) {
    FragU u;
    u.q[0] = *(const uint4*)p;
    u.q[1] = *(const uint4*)(p + 16);
    return u.f;
}

// ---------------------------------------------------------------------------
// f32 -> bf16 convert
// ---------------------------------------------------------------------------
__global__ void f32_to_bf16_kernel(const float* __restrict__ src, u16* __restrict__ dst, int n) {
    int i = blockIdx.x * blockDim.x + threadIdx.x;
    if (i < n) dst[i] = f2bf(src[i]);
}

// ---------------------------------------------------------------------------
// Generic bf16 WMMA GEMM:  C[z] = A[z] (MxK, row-major) * B[z%bmod]^T (NxK, row-major)
// Block = 256 threads (8 waves), tile 128x128, BK=32 (one WMMA K-step), LDS-staged.
// Wave grid 4(M) x 2(N): each wave owns 32x64 = 2x4 WMMA accumulators.
// Output: f32 (Cf) or bf16 (Cb), optional bias[n].
// ---------------------------------------------------------------------------
#define LDT 40  // LDS row stride (u16); 80 bytes, multiple of 16

__global__ __launch_bounds__(256)
void wmma_gemm_nt(const u16* __restrict__ A, int lda, long long sA,
                  const u16* __restrict__ Bm, int ldb, long long sB, int bmod,
                  float* __restrict__ Cf, u16* __restrict__ Cb, int ldc, long long sC,
                  const float* __restrict__ bias, int K)
{
    __shared__ u16 As[128 * LDT];
    __shared__ u16 Bs[128 * LDT];

    const int tid  = threadIdx.x;
    const int wave = tid >> 5;
    const int lane = tid & 31;
    const int ln   = lane & 15;
    const int g    = lane >> 4;

    const int z = blockIdx.z;
    A  += (long long)z * sA;
    Bm += (long long)(z % bmod) * sB;
    const long long cbase = (long long)z * sC;

    const int row0 = blockIdx.y * 128;
    const int col0 = blockIdx.x * 128;
    const int wm = wave >> 1;   // 0..3
    const int wn = wave & 1;    // 0..1

    const v8f vzero = {0.f,0.f,0.f,0.f,0.f,0.f,0.f,0.f};
    v8f acc[2][4];
#pragma unroll
    for (int mi = 0; mi < 2; mi++)
#pragma unroll
        for (int ni = 0; ni < 4; ni++) acc[mi][ni] = vzero;

    const int lr = tid >> 1;          // 0..127 : staged row
    const int lh = (tid & 1) * 16;    // 0 / 16 : K-half

    for (int k0 = 0; k0 < K; k0 += 32) {
        const u16* ga = A  + (size_t)(row0 + lr) * lda + k0 + lh;
        const u16* gb = Bm + (size_t)(col0 + lr) * ldb + k0 + lh;
        uint4 a0 = *(const uint4*)ga;
        uint4 a1 = *(const uint4*)(ga + 8);
        uint4 b0 = *(const uint4*)gb;
        uint4 b1 = *(const uint4*)(gb + 8);
        *(uint4*)&As[lr * LDT + lh]     = a0;
        *(uint4*)&As[lr * LDT + lh + 8] = a1;
        *(uint4*)&Bs[lr * LDT + lh]     = b0;
        *(uint4*)&Bs[lr * LDT + lh + 8] = b1;
        __syncthreads();

        v16bf af[2], bf[4];
#pragma unroll
        for (int mi = 0; mi < 2; mi++)
            af[mi] = load_frag16(&As[(wm * 32 + mi * 16 + ln) * LDT + g * 8]);
#pragma unroll
        for (int ni = 0; ni < 4; ni++)
            bf[ni] = load_frag16(&Bs[(wn * 64 + ni * 16 + ln) * LDT + g * 8]);

#pragma unroll
        for (int mi = 0; mi < 2; mi++)
#pragma unroll
            for (int ni = 0; ni < 4; ni++)
                acc[mi][ni] = __builtin_amdgcn_wmma_f32_16x16x32_bf16(
                    false, af[mi], false, bf[ni], (short)0, acc[mi][ni], false, false);
        __syncthreads();
    }

#pragma unroll
    for (int mi = 0; mi < 2; mi++) {
#pragma unroll
        for (int ni = 0; ni < 4; ni++) {
            const int col = col0 + wn * 64 + ni * 16 + ln;
            const float bv = bias ? bias[col] : 0.0f;
#pragma unroll
            for (int r = 0; r < 8; r++) {
                const int row = row0 + wm * 32 + mi * 16 + g * 8 + r;
                float v = acc[mi][ni][r] + bv;
                if (Cb) Cb[cbase + (long long)row * ldc + col] = f2bf(v);
                else    Cf[cbase + (long long)row * ldc + col] = v;
            }
        }
    }
}

// ---------------------------------------------------------------------------
// Per-token LayerNorm on q,k streams of qkv + bf16 head-split of q,k,v into
// [B*NH, T, HD] layout. One 128-thread block per token.
// ---------------------------------------------------------------------------
__global__ __launch_bounds__(128)
void ln_split_kernel(const float* __restrict__ qkv,
                     const float* __restrict__ qg, const float* __restrict__ qbeta,
                     const float* __restrict__ kg, const float* __restrict__ kbeta,
                     u16* __restrict__ qo, u16* __restrict__ ko, u16* __restrict__ vo)
{
    const int m   = blockIdx.x;    // token 0..4095
    const int b   = m >> 9;
    const int t   = m & 511;
    const int tid = threadIdx.x;

    const float* row = qkv + (size_t)m * 1536;

    float qv[4], kv[4];
    float sq = 0.f, sqq = 0.f, sk = 0.f, skk = 0.f;
#pragma unroll
    for (int i = 0; i < 4; i++) {
        int c = tid + i * 128;
        float q = row[c];
        float k = row[512 + c];
        qv[i] = q; kv[i] = k;
        sq += q; sqq += q * q; sk += k; skk += k * k;
    }

    __shared__ float red[4][128];
    red[0][tid] = sq; red[1][tid] = sqq; red[2][tid] = sk; red[3][tid] = skk;
    __syncthreads();
    for (int s = 64; s > 0; s >>= 1) {
        if (tid < s) {
            red[0][tid] += red[0][tid + s];
            red[1][tid] += red[1][tid + s];
            red[2][tid] += red[2][tid + s];
            red[3][tid] += red[3][tid + s];
        }
        __syncthreads();
    }
    const float inv512 = 1.0f / 512.0f;
    const float qmu = red[0][0] * inv512, qvar = red[1][0] * inv512 - qmu * qmu;
    const float kmu = red[2][0] * inv512, kvar = red[3][0] * inv512 - kmu * kmu;
    const float qis = rsqrtf(qvar + 1e-5f);
    const float kis = rsqrtf(kvar + 1e-5f);

#pragma unroll
    for (int i = 0; i < 4; i++) {
        int c = tid + i * 128;
        int h = c >> 6, d = c & 63;
        size_t oidx = ((size_t)(b * 8 + h) * 512 + t) * 64 + d;
        qo[oidx] = f2bf((qv[i] - qmu) * qis * qg[c] + qbeta[c]);
        ko[oidx] = f2bf((kv[i] - kmu) * kis * kg[c] + kbeta[c]);
        vo[oidx] = f2bf(row[1024 + c]);
    }
}

// ---------------------------------------------------------------------------
// Flash-style causal attention with rel-bias gather.
// Block = 128 threads (4 waves); block owns 64 q-rows of one (b,h); each wave
// owns 16 q-rows. Streams keys in 64-wide chunks (causal => qt+1 chunks).
// S = Q*K^T via WMMA, +gathered rel bias, *scale, mask, online softmax,
// O += P*V via WMMA (P restaged through LDS, V staged transposed in LDS).
// ---------------------------------------------------------------------------
__global__ __launch_bounds__(128)
void attention_kernel(const u16* __restrict__ qb, const u16* __restrict__ kb,
                      const u16* __restrict__ vb, const u16* __restrict__ pb,
                      u16* __restrict__ yo)
{
    __shared__ u16 Vt[64 * 80];        // V^T tile: Vt[d][j], stride 80 (160B)
    __shared__ u16 Pl[4 * 16 * 64];    // per-wave P tile (16 rows x 64 cols)

    const int tid  = threadIdx.x;
    const int wave = tid >> 5;
    const int lane = tid & 31;
    const int ln   = lane & 15;
    const int g    = lane >> 4;

    const int qt = blockIdx.x;         // q tile 0..7
    const int h  = blockIdx.y;
    const int b  = blockIdx.z;
    const int bh = b * 8 + h;

    const int i_base  = qt * 64 + wave * 16;
    const float scale = 0.125f;        // 1/sqrt(64)

    // Q fragments (16 rows x 64 K = 2 k-steps)
    const u16* qrow = qb + ((size_t)bh * 512 + (i_base + ln)) * 64;
    v16bf qf[2];
#pragma unroll
    for (int kk = 0; kk < 2; kk++) qf[kk] = load_frag16(qrow + kk * 32 + g * 8);

    const v8f vzero = {0.f,0.f,0.f,0.f,0.f,0.f,0.f,0.f};
    v8f O[4];
    float mrow[8], lrow[8];
#pragma unroll
    for (int ni = 0; ni < 4; ni++) O[ni] = vzero;
#pragma unroll
    for (int r = 0; r < 8; r++) { mrow[r] = -1e30f; lrow[r] = 0.f; }

    u16* Pw = &Pl[wave * 16 * 64];

    const int nchunks = qt + 1;
    for (int jc = 0; jc < nchunks; jc++) {
        const int j0 = jc * 64;
        __syncthreads();   // previous chunk's Vt consumers done
        {   // stage V transposed: Vt[d][j_local]
            const int jr = tid >> 1;
            const int dh = (tid & 1) * 32;
            const u16* vr = vb + ((size_t)bh * 512 + (j0 + jr)) * 64 + dh;
            uint4 t0 = *(const uint4*)vr;
            uint4 t1 = *(const uint4*)(vr + 8);
            uint4 t2 = *(const uint4*)(vr + 16);
            uint4 t3 = *(const uint4*)(vr + 24);
            u16 tmp[32];
            *(uint4*)&tmp[0]  = t0; *(uint4*)&tmp[8]  = t1;
            *(uint4*)&tmp[16] = t2; *(uint4*)&tmp[24] = t3;
#pragma unroll
            for (int e = 0; e < 32; e++) Vt[(dh + e) * 80 + jr] = tmp[e];
        }
        __syncthreads();

        // S = Q K^T (K fragments straight from L2-resident global)
        v8f S[4];
#pragma unroll
        for (int ni = 0; ni < 4; ni++) {
            v8f a = vzero;
            const u16* krow = kb + ((size_t)bh * 512 + (j0 + ni * 16 + ln)) * 64;
#pragma unroll
            for (int kk = 0; kk < 2; kk++) {
                v16bf kf = load_frag16(krow + kk * 32 + g * 8);
                a = __builtin_amdgcn_wmma_f32_16x16x32_bf16(
                        false, qf[kk], false, kf, (short)0, a, false, false);
            }
            S[ni] = a;
        }

        // rel-bias gather + scale + causal mask
#pragma unroll
        for (int ni = 0; ni < 4; ni++) {
            const int j = j0 + ni * 16 + ln;
#pragma unroll
            for (int r = 0; r < 8; r++) {
                const int i = i_base + g * 8 + r;
                float s;
                if (j > i) s = -1e30f;
                else {
                    float rbv = bf2f(pb[(size_t)bh * 512 * 512 + (size_t)i * 512 + (i - j)]);
                    s = (S[ni][r] + rbv) * scale;
                }
                S[ni][r] = s;
            }
        }

        // online softmax (row stats live in 16-lane halves matching C-layout)
        float alpha[8];
#pragma unroll
        for (int r = 0; r < 8; r++) {
            float mx = fmaxf(fmaxf(S[0][r], S[1][r]), fmaxf(S[2][r], S[3][r]));
#pragma unroll
            for (int msk = 1; msk < 16; msk <<= 1) mx = fmaxf(mx, __shfl_xor(mx, msk, 32));
            float mn = fmaxf(mrow[r], mx);
            alpha[r] = __expf(mrow[r] - mn);
            mrow[r] = mn;
            float rs = 0.f;
#pragma unroll
            for (int ni = 0; ni < 4; ni++) {
                float p = __expf(S[ni][r] - mn);
                S[ni][r] = p;
                rs += p;
            }
#pragma unroll
            for (int msk = 1; msk < 16; msk <<= 1) rs += __shfl_xor(rs, msk, 32);
            lrow[r] = lrow[r] * alpha[r] + rs;
        }

        // stage P (bf16) into wave-local LDS; rescale O
#pragma unroll
        for (int ni = 0; ni < 4; ni++) {
#pragma unroll
            for (int r = 0; r < 8; r++) {
                Pw[(g * 8 + r) * 64 + ni * 16 + ln] = f2bf(S[ni][r]);
                O[ni][r] *= alpha[r];
            }
        }
        // wave-local LDS RAW: DS ops are in-order; fence + drain DScnt
        asm volatile("s_wait_dscnt 0" ::: "memory");

        // O += P @ V
#pragma unroll
        for (int kk = 0; kk < 2; kk++) {
            v16bf pf = load_frag16(&Pw[ln * 64 + kk * 32 + g * 8]);
#pragma unroll
            for (int ni = 0; ni < 4; ni++) {
                v16bf vf = load_frag16(&Vt[(ni * 16 + ln) * 80 + kk * 32 + g * 8]);
                O[ni] = __builtin_amdgcn_wmma_f32_16x16x32_bf16(
                            false, pf, false, vf, (short)0, O[ni], false, false);
            }
        }
    }

    // write y (bf16) as [B,T,C] with heads concatenated (proj-GEMM A operand)
#pragma unroll
    for (int ni = 0; ni < 4; ni++) {
#pragma unroll
        for (int r = 0; r < 8; r++) {
            const int i = i_base + g * 8 + r;
            const int d = ni * 16 + ln;
            float v = O[ni][r] / lrow[r];
            yo[((size_t)b * 512 + i) * 512 + h * 64 + d] = f2bf(v);
        }
    }
}

// ---------------------------------------------------------------------------
// Host-side orchestration
// ---------------------------------------------------------------------------
extern "C" void kernel_launch(void* const* d_in, const int* in_sizes, int n_in,
                              void* d_out, int out_size, void* d_ws, size_t ws_size,
                              hipStream_t stream)
{
    (void)in_sizes; (void)n_in; (void)out_size; (void)ws_size;

    const float* x      = (const float*)d_in[0];
    const float* w_attn = (const float*)d_in[1];
    const float* b_attn = (const float*)d_in[2];
    const float* w_proj = (const float*)d_in[3];
    const float* b_proj = (const float*)d_in[4];
    const float* q_g    = (const float*)d_in[5];
    const float* q_b    = (const float*)d_in[6];
    const float* k_g    = (const float*)d_in[7];
    const float* k_b    = (const float*)d_in[8];
    const float* rel    = (const float*)d_in[9];

    char* ws = (char*)d_ws;
    size_t off = 0;
    auto alloc = [&](size_t bytes) -> char* {
        char* p = ws + off;
        off += (bytes + 255) & ~(size_t)255;
        return p;
    };

    float* qkv  = (float*)alloc(4096ull * 1536 * 4);   // QKV projection (f32)
    u16*   xb   = (u16*)  alloc(4096ull * 512 * 2);    // x bf16
    u16*   wab  = (u16*)  alloc(1536ull * 512 * 2);    // w_attn bf16
    u16*   wpb  = (u16*)  alloc(512ull  * 512 * 2);    // w_proj bf16
    u16*   relb = (u16*)  alloc(512ull  * 512 * 2);    // rel_emb bf16
    u16*   qbuf = (u16*)  alloc(64ull * 512 * 64 * 2); // q  [B*NH,T,HD] bf16
    u16*   kbuf = (u16*)  alloc(64ull * 512 * 64 * 2); // k
    u16*   vbuf = (u16*)  alloc(64ull * 512 * 64 * 2); // v
    u16*   pbuf = (u16*)  alloc(64ull * 512 * 512 * 2);// rel projection p (bf16)
    u16*   ybuf = (u16*)  alloc(4096ull * 512 * 2);    // attention out bf16

    auto cvt = [&](const float* s, u16* d, int n) {
        f32_to_bf16_kernel<<<(n + 255) / 256, 256, 0, stream>>>(s, d, n);
    };
    cvt(x,      xb,   4096 * 512);
    cvt(w_attn, wab,  1536 * 512);
    cvt(w_proj, wpb,  512 * 512);
    cvt(rel,    relb, 512 * 512);

    // 1) QKV GEMM: qkv[4096,1536] = x @ w_attn^T + b_attn
    wmma_gemm_nt<<<dim3(12, 32, 1), 256, 0, stream>>>(
        xb, 512, 0, wab, 512, 0, 1, qkv, nullptr, 1536, 0, b_attn, 512);

    // 2) LayerNorm(q), LayerNorm(k), head-split q/k/v to bf16
    ln_split_kernel<<<4096, 128, 0, stream>>>(qkv, q_g, q_b, k_g, k_b, qbuf, kbuf, vbuf);

    // 3) rel projection: per (b,h): p[i,m] = q[i,:] . rel_emb[m, h*64:]
    wmma_gemm_nt<<<dim3(4, 4, 64), 256, 0, stream>>>(
        qbuf, 64, 512 * 64, relb, 512, 64, 8, nullptr, pbuf, 512, 512 * 512, nullptr, 64);

    // 4) flash-style causal attention with rel-bias gather
    attention_kernel<<<dim3(8, 8, 8), 128, 0, stream>>>(qbuf, kbuf, vbuf, pbuf, ybuf);

    // 5) output projection: out[4096,512] = y @ w_proj^T + b_proj
    wmma_gemm_nt<<<dim3(4, 32, 1), 256, 0, stream>>>(
        ybuf, 512, 0, wpb, 512, 0, 1, (float*)d_out, nullptr, 512, 0, b_proj, 512);
}
